// S4_55860344652355
// MI455X (gfx1250) — compile-verified, matrix operand order checked by reading, not run on hardware
//
#include <hip/hip_runtime.h>
#include <hip/hip_bf16.h>

// Problem constants (match reference)
#define LY 4
#define HD 512      // d_model
#define NS 32       // complex state modes == wave32 lanes (one mode per lane)
#define BB 8
#define LL 4096
#define BL (BB * LL)   // 32768 tokens

typedef __attribute__((ext_vector_type(16))) __bf16 v16bf;
typedef __attribute__((ext_vector_type(8)))  __bf16 v8bf;
typedef __attribute__((ext_vector_type(8)))  float  v8f;
typedef __attribute__((ext_vector_type(4)))  unsigned int u32x4;
typedef __attribute__((ext_vector_type(8)))  int i32x8;
typedef __attribute__((ext_vector_type(4)))  int i32x4;

// TDM availability (device pass only; falls back to load+ds_store otherwise)
#if defined(__AMDGCN__) && __has_builtin(__builtin_amdgcn_tensor_load_to_lds)
#define HAVE_TDM 1
#endif

// ---------------------------------------------------------------------------
// Small helpers
// ---------------------------------------------------------------------------
__device__ __forceinline__ float warp_sum32(float v) {
#pragma unroll
  for (int off = 16; off > 0; off >>= 1) v += __shfl_xor(v, off, 32);
  return v;
}

__device__ __forceinline__ float gelu_tanh(float y) {
  // jax.nn.gelu default (approximate=True)
  const float c0 = 0.7978845608028654f;   // sqrt(2/pi)
  const float c1 = 0.044715f;
  float t = tanhf(c0 * (y + c1 * y * y * y));
  return 0.5f * y * (1.0f + t);
}

// Build a 16-element bf16 WMMA fragment from LDS per ISA §7.12.2:
// VGPRs 0-3 hold K = kh*8+0..7 (at p), VGPRs 4-7 hold K = 16+kh*8+0..7 (p+16).
__device__ __forceinline__ v16bf ld_frag(const __bf16* p) {
  v8bf lo = *(const v8bf*)p;
  v8bf hi = *(const v8bf*)(p + 16);
  return __builtin_shufflevector(lo, hi, 0, 1, 2, 3, 4, 5, 6, 7,
                                         8, 9, 10, 11, 12, 13, 14, 15);
}

// ---------------------------------------------------------------------------
// 0) Convert Wout (all layers) to bf16 once per call
// ---------------------------------------------------------------------------
__global__ __launch_bounds__(256) void cvt_bf16_kernel(const float* __restrict__ src,
                                                       __bf16* __restrict__ dst, int n) {
  int i = blockIdx.x * blockDim.x + threadIdx.x;
  if (i < n) dst[i] = (__bf16)src[i];
}

// ---------------------------------------------------------------------------
// 1) Per-layer discretization: w = exp(dt*A), Cd = C*(exp(dt*A)-1)/A   (H,N)
// ---------------------------------------------------------------------------
__global__ __launch_bounds__(256) void coeff_kernel(
    const float* __restrict__ log_dt, const float* __restrict__ A_re,
    const float* __restrict__ A_im,   const float* __restrict__ C_re,
    const float* __restrict__ C_im,
    float* __restrict__ wre, float* __restrict__ wim,
    float* __restrict__ cre, float* __restrict__ cim) {
  int idx = blockIdx.x * blockDim.x + threadIdx.x;
  if (idx >= HD * NS) return;
  int h = idx / NS;
  float dt = expf(log_dt[h]);
  float ar = A_re[idx], ai = A_im[idx];
  float dr = dt * ar, di = dt * ai;
  float em = expf(dr);
  float er = em * cosf(di), ei = em * sinf(di);       // exp(dt*A)
  wre[idx] = er; wim[idx] = ei;
  float inv = 1.0f / (ar * ar + ai * ai);
  float gr = er - 1.0f;
  float fr = (gr * ar + ei * ai) * inv;               // (E-1)/A
  float fi = (ei * ar - gr * ai) * inv;
  float cr = C_re[idx], ci = C_im[idx];
  cre[idx] = cr * fr - ci * fi;                       // Cd = C*(E-1)/A
  cim[idx] = cr * fi + ci * fr;
}

// ---------------------------------------------------------------------------
// 2) LayerNorm: x (B*L, H) -> z (B*L, H).  128 threads / row, 4 elems / thread.
// ---------------------------------------------------------------------------
__global__ __launch_bounds__(128) void ln_kernel(const float* __restrict__ x,
                                                 const float* __restrict__ lw,
                                                 const float* __restrict__ lb,
                                                 float* __restrict__ z) {
  __shared__ float smem[8];
  int row = blockIdx.x;
  int t = threadIdx.x;
  const float* xp = x + (size_t)row * HD;
  float v[4];
  float s = 0.0f;
#pragma unroll
  for (int i = 0; i < 4; ++i) { v[i] = xp[t + i * 128]; s += v[i]; }
  s = warp_sum32(s);
  if ((t & 31) == 0) smem[t >> 5] = s;
  __syncthreads();
  float mean = (smem[0] + smem[1] + smem[2] + smem[3]) * (1.0f / HD);
  float vs = 0.0f;
#pragma unroll
  for (int i = 0; i < 4; ++i) { float d = v[i] - mean; vs += d * d; }
  vs = warp_sum32(vs);
  if ((t & 31) == 0) smem[4 + (t >> 5)] = vs;
  __syncthreads();
  float var = (smem[4] + smem[5] + smem[6] + smem[7]) * (1.0f / HD);
  float rinv = rsqrtf(var + 1e-5f);
  float* zp = z + (size_t)row * HD;
#pragma unroll
  for (int i = 0; i < 4; ++i) {
    int c = t + i * 128;
    zp[c] = (v[i] - mean) * rinv * lw[c] + lb[c];
  }
}

// ---------------------------------------------------------------------------
// 3) SSM recurrence (replaces the FFT conv): one wave per (b,h), lane = mode n.
//    s <- w*s + u;  y = 2*Re(sum_n Cd_n * s_n) + D*u;  store gelu(y) as bf16.
//    z is (B,L,H); per-step u is wave-uniform and served from L2 (z fits in
//    the 192MB L2).  Cross-lane sum = 5-step shfl_xor butterfly (wave32).
// ---------------------------------------------------------------------------
__global__ __launch_bounds__(256) void s4_scan_kernel(
    const float* __restrict__ z,                     // (B,L,H)
    const float* __restrict__ wre, const float* __restrict__ wim,  // (H,N)
    const float* __restrict__ cre, const float* __restrict__ cim,  // (H,N)
    const float* __restrict__ Dv,                    // (H)
    __bf16* __restrict__ g)                          // (B,L,H) gelu output
{
  int wv = (blockIdx.x * blockDim.x + threadIdx.x) >> 5;
  wv = __builtin_amdgcn_readfirstlane(wv);           // force SGPR-uniform
  int lane = threadIdx.x & 31;
  int b = wv / HD;
  int h = wv % HD;
  if (b >= BB) return;

  float wr = wre[h * NS + lane], wi = wim[h * NS + lane];
  float cr = cre[h * NS + lane], ci = cim[h * NS + lane];
  float d  = Dv[h];

  const float* up = z + (size_t)b * LL * HD + h;
  __bf16*      gp = g + (size_t)b * LL * HD + h;

  float sr = 0.0f, si = 0.0f;
#pragma unroll 4
  for (int l = 0; l < LL; ++l) {
    float u = up[(size_t)l * HD];                    // uniform scalar load
    float nsr = fmaf(wr, sr, fmaf(-wi, si, u));      // s = w*s + u (complex)
    float nsi = fmaf(wr, si, wi * sr);
    sr = nsr; si = nsi;
    float part = fmaf(cr, sr, -ci * si);             // Re(Cd * s)
    part = warp_sum32(part);
    float y = 2.0f * part + d * u;
    if (lane == 0) gp[(size_t)l * HD] = (__bf16)gelu_tanh(y);
  }
}

// ---------------------------------------------------------------------------
// 4) Fused GEMM (bf16 WMMA, f32 acc) + bias + GLU + residual.
//    P = G @ Wout + bout ; out += P[:, :H] * sigmoid(P[:, H:])
//    Block: 256 threads = 8 waves.  BM=256 rows, 16 GLU column pairs.
//    Each wave owns TWO 16-row A fragments and both GLU halves:
//    4 WMMAs per K-step with full B-fragment reuse.
//    A tile staged global->LDS by the Tensor Data Mover when available.
// ---------------------------------------------------------------------------
#define BM 256
#define BK 32
__global__ __launch_bounds__(256) void gemm_glu_res_kernel(
    const __bf16* __restrict__ g,     // (B*L, H)  gelu activations
    const __bf16* __restrict__ wb,    // (H, 2H)   layer weights, bf16
    const float*  __restrict__ bout,  // (2H)
    float* __restrict__ xio)          // (B*L, H)  residual in/out
{
  __shared__ __bf16 sa[BM * BK];        // 256x32 A tile           (16 KB)
  __shared__ __bf16 sb0[16 * BK];       // 16 cols x 32 k, half 0, transposed
  __shared__ __bf16 sb1[16 * BK];       // half 1

  const int m0 = blockIdx.x * BM;
  const int n0 = blockIdx.y * 16;       // column within each GLU half
  const int tid = threadIdx.x;
  const int waveId = tid >> 5;
  const int lane = tid & 31;
  const int r  = lane & 15;             // fragment row/col within 16
  const int kh = lane >> 4;             // K-half select per ISA layout

  v8f c00 = {}, c01 = {}, c10 = {}, c11 = {};

  for (int k0 = 0; k0 < HD; k0 += BK) {
    // ---- stage A tile (256 rows x 32 k, bf16) into LDS ----
#if defined(HAVE_TDM)
    if (tid < 32) {                     // one TDM descriptor, issued by wave 0
      unsigned long long ga =
          (unsigned long long)(uintptr_t)(g + (size_t)m0 * HD + k0);
      unsigned lds = (unsigned)(uintptr_t)&sa[0];
      // D# group 0: count=1, lds_addr, global_addr[56:0], type=2 ("image")
      u32x4 g0;
      g0[0] = 1u;
      g0[1] = lds;
      g0[2] = (unsigned)(ga & 0xFFFFFFFFu);
      g0[3] = (unsigned)((ga >> 32) & 0x1FFFFFFu) | (2u << 30);
      // D# group 1: data_size=2B; tensor 512 x 32768, row stride 512 elems;
      //             tile 32 x 256 (k-major within a row)
      const unsigned td0 = HD, td1 = (unsigned)BL, t0 = BK, t1 = BM;
      const unsigned long long s0 = HD;
      i32x8 g1;
      g1[0] = (int)(1u << 16);                                   // data_size=1 -> 2B
      g1[1] = (int)((td0 & 0xFFFFu) << 16);                      // tensor_dim0 lo
      g1[2] = (int)((td0 >> 16) | ((td1 & 0xFFFFu) << 16));      // td0 hi | td1 lo
      g1[3] = (int)((td1 >> 16) | ((t0 & 0xFFFFu) << 16));       // td1 hi | tile0
      g1[4] = (int)(t1 & 0xFFFFu);                               // tile1 (tile2=0)
      g1[5] = (int)(s0 & 0xFFFFFFFFull);                         // dim0 stride lo
      g1[6] = (int)((s0 >> 32) & 0xFFFFull);                     // stride hi
      g1[7] = 0;
      i32x4 zz = {0, 0, 0, 0};                                   // 2D: groups 2/3 unused
#if defined(__clang_major__) && (__clang_major__ >= 23)
      i32x8 zz8 = {0, 0, 0, 0, 0, 0, 0, 0};
      __builtin_amdgcn_tensor_load_to_lds(g0, g1, zz, zz, zz8, 0);
#else
      __builtin_amdgcn_tensor_load_to_lds(g0, g1, zz, zz, 0);
#endif
      __builtin_amdgcn_s_wait_tensorcnt(0);
    }
#else
    {
      // fallback: each thread copies one 32-elem row (4x 16B vectors)
      const __bf16* asrc = g + (size_t)(m0 + tid) * HD + k0;
      v8bf* adst = (v8bf*)&sa[tid * BK];
      adst[0] = ((const v8bf*)asrc)[0];
      adst[1] = ((const v8bf*)asrc)[1];
      adst[2] = ((const v8bf*)asrc)[2];
      adst[3] = ((const v8bf*)asrc)[3];
      if (k0 + BK < HD) __builtin_prefetch(asrc + BK, 0, 1);  // global_prefetch_b8
    }
#endif
    // ---- stage B tiles transposed: sb[c*BK + k] = W[k0+k][half*H + n0+c] ----
#pragma unroll
    for (int e = 0; e < 2; ++e) {                    // statically unrolled
      int idx = tid + e * 256;
      int kk = idx >> 4, cc = idx & 15;
      const __bf16* wsrc = wb + (size_t)(k0 + kk) * (2 * HD) + n0 + cc;
      sb0[cc * BK + kk] = wsrc[0];
      sb1[cc * BK + kk] = wsrc[HD];
    }
    if (k0 + BK < HD) {
      const __bf16* wnext =
          wb + (size_t)(k0 + BK + (tid >> 4)) * (2 * HD) + n0 + (tid & 15);
      __builtin_prefetch(wnext, 0, 1);               // global_prefetch_b8
    }
    __syncthreads();

    // ---- fragments (ISA §7.12.2 16-bit layouts) + 4 WMMAs ----
    v16bf af0 = ld_frag(&sa[(waveId * 32 + r) * BK + kh * 8]);
    v16bf af1 = ld_frag(&sa[(waveId * 32 + 16 + r) * BK + kh * 8]);
    v16bf bf0 = ld_frag(&sb0[r * BK + kh * 8]);
    v16bf bf1 = ld_frag(&sb1[r * BK + kh * 8]);

    c00 = __builtin_amdgcn_wmma_f32_16x16x32_bf16(false, af0, false, bf0,
                                                  (short)0, c00, false, false);
    c01 = __builtin_amdgcn_wmma_f32_16x16x32_bf16(false, af0, false, bf1,
                                                  (short)0, c01, false, false);
    c10 = __builtin_amdgcn_wmma_f32_16x16x32_bf16(false, af1, false, bf0,
                                                  (short)0, c10, false, false);
    c11 = __builtin_amdgcn_wmma_f32_16x16x32_bf16(false, af1, false, bf1,
                                                  (short)0, c11, false, false);
    __syncthreads();
  }

  // ---- epilogue: bias + GLU + residual.  C/D layout: VGPR e holds
  //      M = e + (lane>=16 ? 8 : 0), N = lane%16. ----
  const float b0v = bout[n0 + r];
  const float b1v = bout[HD + n0 + r];
#pragma unroll
  for (int e = 0; e < 8; ++e) {
    int mm = waveId * 32 + e + kh * 8;
    float p0 = c00[e] + b0v;
    float p1 = c01[e] + b1v;
    xio[(size_t)(m0 + mm) * HD + n0 + r] += p0 * (1.0f / (1.0f + expf(-p1)));
    p0 = c10[e] + b0v;
    p1 = c11[e] + b1v;
    xio[(size_t)(m0 + mm + 16) * HD + n0 + r] += p0 * (1.0f / (1.0f + expf(-p1)));
  }
}

// ---------------------------------------------------------------------------
// Host-side orchestration
// ---------------------------------------------------------------------------
extern "C" void kernel_launch(void* const* d_in, const int* in_sizes, int n_in,
                              void* d_out, int out_size, void* d_ws, size_t ws_size,
                              hipStream_t stream) {
  (void)in_sizes; (void)n_in; (void)out_size; (void)ws_size;
  const float* x      = (const float*)d_in[0];   // (B,L,H)
  const float* log_dt = (const float*)d_in[1];   // (LY,H)
  const float* A_re   = (const float*)d_in[2];   // (LY,H,N)
  const float* A_im   = (const float*)d_in[3];
  const float* C_re   = (const float*)d_in[4];
  const float* C_im   = (const float*)d_in[5];
  const float* Dm     = (const float*)d_in[6];   // (LY,H)
  const float* ln_w   = (const float*)d_in[7];   // (LY,H)
  const float* ln_b   = (const float*)d_in[8];
  const float* Wout   = (const float*)d_in[9];   // (LY,H,2H)
  const float* bout   = (const float*)d_in[10];  // (LY,2H)
  float* out = (float*)d_out;                    // (B,L,H) residual stream

  // Workspace layout
  char* wsb = (char*)d_ws;
  const size_t ZB = (size_t)BL * HD * sizeof(float);    // 64 MB  z
  const size_t GB = (size_t)BL * HD * sizeof(__bf16);   // 32 MB  gelu acts
  const size_t WB = (size_t)LY * HD * 2 * HD * sizeof(__bf16); // 4 MB weights
  float*  z     = (float*)wsb;
  __bf16* g     = (__bf16*)(wsb + ZB);
  __bf16* woutb = (__bf16*)(wsb + ZB + GB);
  float*  wre   = (float*)(wsb + ZB + GB + WB);
  float*  wim   = wre + HD * NS;
  float*  cre   = wim + HD * NS;
  float*  cim   = cre + HD * NS;

  // residual stream lives in d_out
  hipMemcpyAsync(out, x, (size_t)BL * HD * sizeof(float),
                 hipMemcpyDeviceToDevice, stream);

  // convert all Wout layers to bf16 once
  {
    int n = LY * HD * 2 * HD;
    cvt_bf16_kernel<<<(n + 255) / 256, 256, 0, stream>>>(Wout, woutb, n);
  }

  for (int i = 0; i < LY; ++i) {
    coeff_kernel<<<(HD * NS + 255) / 256, 256, 0, stream>>>(
        log_dt + (size_t)i * HD,
        A_re + (size_t)i * HD * NS, A_im + (size_t)i * HD * NS,
        C_re + (size_t)i * HD * NS, C_im + (size_t)i * HD * NS,
        wre, wim, cre, cim);

    ln_kernel<<<BL, 128, 0, stream>>>(out, ln_w + (size_t)i * HD,
                                      ln_b + (size_t)i * HD, z);

    // B*H = 4096 waves, 8 waves (256 threads) per block
    s4_scan_kernel<<<(BB * HD) / 8, 256, 0, stream>>>(
        z, wre, wim, cre, cim, Dm + (size_t)i * HD, g);

    dim3 grid(BL / BM, HD / 16);
    gemm_glu_res_kernel<<<grid, 256, 0, stream>>>(
        g, woutb + (size_t)i * HD * 2 * HD, bout + (size_t)i * 2 * HD, out);
  }
}
